// GEncoder_69398081569223
// MI455X (gfx1250) — compile-verified, matrix operand order checked by reading, not run on hardware
//
#include <hip/hip_runtime.h>

typedef __attribute__((ext_vector_type(16))) _Float16 v16h;
typedef __attribute__((ext_vector_type(8)))  float    v8f;

union Frag16 { v16h v; _Float16 h[16]; };

// -------------------------------------------------------------------------
// WMMA GEMM: out[nrows, NOUT] = (RELU_IN ? relu(A) : A)[nrows, K] @ W[K, NOUT]
// f32 in/out, f16 WMMA with f32 accumulate. 128 rows per block (8 waves).
// A-fragment lane layout (16-bit A, 16x32, wave32):
//   lanes 0-15 : row M=lane, elems 0..7 -> K=kt*32+0..7,  elems 8..15 -> K=kt*32+16..23
//   lanes 16-31: row M=lane-16, elems 0..7 -> K=kt*32+8..15, elems 8..15 -> K=kt*32+24..31
// B-fragment mirrors with column N per lane (W staged transposed in LDS).
// C/D: elem r -> row M = r + 8*(lane>>4), col N = lane&15.
// -------------------------------------------------------------------------
template<int K, int NOUT, bool RELU_IN>
__launch_bounds__(256)
__global__ void gemm_wmma_f16(const float* __restrict__ A,
                              const float* __restrict__ W,
                              float* __restrict__ out, int nrows)
{
    constexpr int LDA = K + 8;   // +16B pad: kills 16-way bank conflicts
    constexpr int LDB = K + 8;
    __shared__ _Float16 lA[128 * LDA];
    __shared__ _Float16 lB[NOUT * LDB];

    const int rowbase = blockIdx.x * 128;
    const bool fulltile = (rowbase + 128) <= nrows;

    // Stage A tile (f32 -> f16, optional fused ReLU)
    for (int i = threadIdx.x; i < 128 * K; i += 256) {
        int r = i / K, c = i - r * K;
        int gr = rowbase + r;
        float v = (gr < nrows) ? A[(size_t)gr * K + c] : 0.0f;
        if (RELU_IN) v = fmaxf(v, 0.0f);
        lA[r * LDA + c] = (_Float16)v;
    }
    // Stage W transposed: lB[n][k] = W[k][n]
    for (int i = threadIdx.x; i < NOUT * K; i += 256) {
        int n = i / K, k = i - n * K;
        lB[n * LDB + k] = (_Float16)W[(size_t)k * NOUT + n];
    }
    __syncthreads();

    const int wave = threadIdx.x >> 5;
    const int lane = threadIdx.x & 31;
    const int lrow = lane & 15;
    const int ksel = (lane >> 4) * 8;

    // Hoist all A fragments for this wave's 16-row tile
    Frag16 af[K / 32];
#pragma unroll
    for (int kt = 0; kt < K / 32; ++kt) {
        const _Float16* p = &lA[(wave * 16 + lrow) * LDA + kt * 32 + ksel];
#pragma unroll
        for (int e = 0; e < 8; ++e) { af[kt].h[e] = p[e]; af[kt].h[8 + e] = p[16 + e]; }
    }

    const int rbase = rowbase + wave * 16 + (lane >> 4) * 8;

#pragma unroll
    for (int nt = 0; nt < NOUT / 16; ++nt) {
        v8f acc = {};
#pragma unroll
        for (int kt = 0; kt < K / 32; ++kt) {
            Frag16 bf;
            const _Float16* p = &lB[(nt * 16 + lrow) * LDB + kt * 32 + ksel];
#pragma unroll
            for (int e = 0; e < 8; ++e) { bf.h[e] = p[e]; bf.h[8 + e] = p[16 + e]; }
            acc = __builtin_amdgcn_wmma_f32_16x16x32_f16(
                false, af[kt].v, false, bf.v, (short)0, acc, false, false);
        }
        const int col = nt * 16 + lrow;
        float* op = out + (size_t)rbase * NOUT + col;
        if (fulltile) {
#pragma unroll
            for (int r = 0; r < 8; ++r) op[(size_t)r * NOUT] = acc[r];
        } else {
#pragma unroll
            for (int r = 0; r < 8; ++r)
                if (rbase + r < nrows) op[(size_t)r * NOUT] = acc[r];
        }
    }
}

// -------------------------------------------------------------------------
// Heads: mu = A@Wmu + bmu, logvar = A@Wlv + blv  (A: [nrows,32])
// Staged as one 32x64 LDS B panel (cols 0-31 = Wmu, 32-63 = Wlv).
// out layout: [mu (nrows*32) | logvar (nrows*32)]
// -------------------------------------------------------------------------
__launch_bounds__(256)
__global__ void heads_wmma_f16(const float* __restrict__ A,
                               const float* __restrict__ Wmu, const float* __restrict__ bmu,
                               const float* __restrict__ Wlv, const float* __restrict__ blv,
                               float* __restrict__ out, int nrows)
{
    constexpr int K = 32, LDA = K + 8, LDB = K + 8;
    __shared__ _Float16 lA[128 * LDA];
    __shared__ _Float16 lB[64 * LDB];
    __shared__ float    lbias[64];   // bmu | blv

    const int rowbase = blockIdx.x * 128;
    const bool fulltile = (rowbase + 128) <= nrows;

    for (int i = threadIdx.x; i < 128 * K; i += 256) {
        int r = i / K, c = i - r * K;
        int gr = rowbase + r;
        lA[r * LDA + c] = (_Float16)((gr < nrows) ? A[(size_t)gr * K + c] : 0.0f);
    }
    for (int i = threadIdx.x; i < 64 * K; i += 256) {
        int n = i / K, k = i - n * K;
        float w = (n < 32) ? Wmu[(size_t)k * 32 + n] : Wlv[(size_t)k * 32 + (n - 32)];
        lB[n * LDB + k] = (_Float16)w;
    }
    if (threadIdx.x < 64)
        lbias[threadIdx.x] = (threadIdx.x < 32) ? bmu[threadIdx.x] : blv[threadIdx.x - 32];
    __syncthreads();

    const int wave = threadIdx.x >> 5;
    const int lane = threadIdx.x & 31;
    const int lrow = lane & 15;
    const int ksel = (lane >> 4) * 8;

    Frag16 af;
    {
        const _Float16* p = &lA[(wave * 16 + lrow) * LDA + ksel];
#pragma unroll
        for (int e = 0; e < 8; ++e) { af.h[e] = p[e]; af.h[8 + e] = p[16 + e]; }
    }

    const int rbase = rowbase + wave * 16 + (lane >> 4) * 8;

#pragma unroll
    for (int nt = 0; nt < 4; ++nt) {
        Frag16 bf;
        const _Float16* p = &lB[(nt * 16 + lrow) * LDB + ksel];
#pragma unroll
        for (int e = 0; e < 8; ++e) { bf.h[e] = p[e]; bf.h[8 + e] = p[16 + e]; }
        v8f acc = {};
        acc = __builtin_amdgcn_wmma_f32_16x16x32_f16(
            false, af.v, false, bf.v, (short)0, acc, false, false);

        const int col   = nt * 16 + lrow;
        const float bias = lbias[col];
        // mu block at 0, logvar block at nrows*32; branchless column offset
        const size_t colOff = (col < 32) ? (size_t)col
                                         : (size_t)nrows * 32 + (size_t)(col - 32);
        float* op = out + colOff + (size_t)rbase * 32;
        if (fulltile) {
#pragma unroll
            for (int r = 0; r < 8; ++r) op[(size_t)r * 32] = acc[r] + bias;
        } else {
#pragma unroll
            for (int r = 0; r < 8; ++r)
                if (rbase + r < nrows) op[(size_t)r * 32] = acc[r] + bias;
        }
    }
}

// ------------------------- graph-aggregation kernels ----------------------

__global__ void k_fill_one(float* __restrict__ deg, int n) {
    int i = blockIdx.x * blockDim.x + threadIdx.x;
    if (i < n) deg[i] = 1.0f;  // self-loop contribution
}

__global__ void k_deg_accum(const int* __restrict__ dst, float* __restrict__ deg, int e) {
    int i = blockIdx.x * blockDim.x + threadIdx.x;
    if (i < e) atomicAdd(&deg[dst[i]], 1.0f);
}

__global__ void k_rsqrt_inplace(float* __restrict__ d, int n) {
    int i = blockIdx.x * blockDim.x + threadIdx.x;
    if (i < n) d[i] = rsqrtf(d[i]);  // deg >= 1 always (self-loops)
}

// agg[n][f] = bias[f] + h[n][f] * dinv[n]^2 (self-loop folded in); float4 wide.
__global__ void k_agg_init4(const float4* __restrict__ h4, const float* __restrict__ dinv,
                            const float4* __restrict__ bias4, float4* __restrict__ agg4,
                            int n, int F4) {
    size_t i = (size_t)blockIdx.x * blockDim.x + threadIdx.x;
    if (i >= (size_t)n * F4) return;
    int node = (int)(i / F4);
    int f4   = (int)(i - (size_t)node * F4);
    float di = dinv[node];
    float s  = di * di;
    float4 hv = h4[i];
    float4 b  = bias4[f4];
    float4 o;
    o.x = b.x + hv.x * s; o.y = b.y + hv.y * s;
    o.z = b.z + hv.z * s; o.w = b.w + hv.w * s;
    agg4[i] = o;
}

// One wave per edge; lane-per-feature coalesced gather + f32 global atomics.
// Edge-index stream prefetched ahead via global_prefetch (speculative, drop-safe).
__global__ void k_agg_edges(const int* __restrict__ src, const int* __restrict__ dst,
                            const float* __restrict__ dinv,
                            const float* __restrict__ h, float* __restrict__ agg,
                            int e, int F) {
    int gw   = (blockIdx.x * blockDim.x + threadIdx.x) >> 5;
    int lane = threadIdx.x & 31;
    if (gw >= e) return;
    if (lane == 0) {
        __builtin_prefetch(src + gw + 4096, 0, 0);
        __builtin_prefetch(dst + gw + 4096, 0, 0);
    }
    int s = src[gw], d = dst[gw];
    float nrm = dinv[s] * dinv[d];
    const float* hs = h + (size_t)s * F;
    float* ad = agg + (size_t)d * F;
    for (int f = lane; f < F; f += 32)
        atomicAdd(&ad[f], hs[f] * nrm);
}

// --------------------------------- launch ---------------------------------

extern "C" void kernel_launch(void* const* d_in, const int* in_sizes, int n_in,
                              void* d_out, int out_size, void* d_ws, size_t ws_size,
                              hipStream_t stream) {
    const float* x   = (const float*)d_in[0];
    const int*   ei  = (const int*)d_in[1];
    const float* W1  = (const float*)d_in[2];
    const float* b1  = (const float*)d_in[3];
    const float* W2  = (const float*)d_in[4];
    const float* b2  = (const float*)d_in[5];
    const float* Wmu = (const float*)d_in[6];
    const float* bmu = (const float*)d_in[7];
    const float* Wlv = (const float*)d_in[8];
    const float* blv = (const float*)d_in[9];

    const int N = in_sizes[0] / 128;
    const int E = in_sizes[1] / 2;
    const int* src = ei;
    const int* dst = ei + E;

    float* ws   = (float*)d_ws;
    float* dinv = ws;                           // [N]
    float* bufA = ws + N;                       // h1 [N,64], later h2 [N,32]
    float* bufB = bufA + (size_t)N * 64;        // agg1 [N,64], later agg2 [N,32]
    float* out  = (float*)d_out;

    const int T = 256;
    // degrees -> dinv
    k_fill_one<<<(N + T - 1) / T, T, 0, stream>>>(dinv, N);
    k_deg_accum<<<(E + T - 1) / T, T, 0, stream>>>(dst, dinv, E);
    k_rsqrt_inplace<<<(N + T - 1) / T, T, 0, stream>>>(dinv, N);

    // layer 1: h1 = x @ W1 ; agg1 = b1 + scatter(norm * h1[src]) ; (relu fused into gemm2)
    gemm_wmma_f16<128, 64, false><<<(N + 127) / 128, T, 0, stream>>>(x, W1, bufA, N);
    k_agg_init4<<<(int)(((size_t)N * 16 + T - 1) / T), T, 0, stream>>>(
        (const float4*)bufA, dinv, (const float4*)b1, (float4*)bufB, N, 16);
    k_agg_edges<<<(E + 7) / 8, T, 0, stream>>>(src, dst, dinv, bufA, bufB, E, 64);

    // layer 2: h2 = relu(agg1) @ W2 ; agg2 = b2 + scatter(norm * h2[src])
    gemm_wmma_f16<64, 32, true><<<(N + 127) / 128, T, 0, stream>>>(bufB, W2, bufA, N);
    k_agg_init4<<<(int)(((size_t)N * 8 + T - 1) / T), T, 0, stream>>>(
        (const float4*)bufA, dinv, (const float4*)b2, (float4*)bufB, N, 8);
    k_agg_edges<<<(E + 7) / 8, T, 0, stream>>>(src, dst, dinv, bufA, bufB, E, 32);

    // heads: mu / logvar
    heads_wmma_f16<<<(N + 127) / 128, T, 0, stream>>>(bufB, Wmu, bmu, Wlv, blv, out, N);
}
